// QuantumNeuralNetwork_8555574854208
// MI455X (gfx1250) — matrix operand order repaired and checked by reading, not compiled, not run on hardware
//
#include <hip/hip_runtime.h>
#include <hip/hip_bf16.h>
#include <math.h>

typedef __attribute__((ext_vector_type(16))) _Float16 v16h;
typedef __attribute__((ext_vector_type(8)))  _Float16 v8h;
typedef __attribute__((ext_vector_type(8)))  float    v8f;

#define BM 128
#define BN 128
#define BK 32
#define BKP 40   // padded LDS row stride in halves (80B rows -> conflict-free b128 reads)

enum { EPI_F16 = 0, EPI_BIAS_QACT = 1, EPI_F32 = 2 };

__device__ __forceinline__ float qact(float x) {
  // 0.4*sigmoid(x) + 0.3*tanh(x) + 0.3*relu(x)
  float s = 1.0f / (1.0f + __expf(-x));
  float t = tanhf(x);
  float r = fmaxf(x, 0.0f);
  return 0.4f * s + 0.3f * t + 0.3f * r;
}

// ---------------------------------------------------------------------------
// GEMM: C[M,N] = A[M,K] * Bt[N,K]^T   (A row-major, B stored transposed [N,K])
// f16 inputs, f32 WMMA accumulation. 256 threads = 8 waves; each wave owns a
// 32x64 output tile (2x4 v_wmma_f32_16x16x32_f16 accumulators).
// Double-buffered LDS: one barrier per K-tile, global loads 2 tiles ahead.
// ---------------------------------------------------------------------------
template <int EPI>
__global__ __launch_bounds__(256) void k_gemm(const _Float16* __restrict__ A,
                                              const _Float16* __restrict__ Bt,
                                              const float* __restrict__ bias,
                                              _Float16* __restrict__ C16,
                                              float* __restrict__ C32,
                                              int M, int N, int K) {
  __shared__ _Float16 sA[2][BM * BKP];
  __shared__ _Float16 sB[2][BN * BKP];

  const int tid  = threadIdx.x;
  const int lane = tid & 31;   // wave32
  const int wave = tid >> 5;   // 0..7
  const int wm   = wave & 3;   // 4 waves along M
  const int wn   = wave >> 2;  // 2 waves along N
  const int half = lane >> 4;  // lane-half select for fragment K split
  const int l16  = lane & 15;

  const int bMr = blockIdx.y * BM;
  const int bNr = blockIdx.x * BN;

  // global->LDS: each thread moves one 16-half (32B) chunk per tile
  const int ldRow = tid >> 1;
  const int ldK   = (tid & 1) * 16;
  const int ldOff = ldRow * BKP + ldK;

  const _Float16* gA = A  + (size_t)(bMr + ldRow) * K + ldK;
  const _Float16* gB = Bt + (size_t)(bNr + ldRow) * K + ldK;

  v8f acc[2][4];
#pragma unroll
  for (int i = 0; i < 2; ++i)
#pragma unroll
    for (int j = 0; j < 4; ++j) acc[i][j] = (v8f){};

  const int KT = K / BK;

  // Prolog: tile 0 -> LDS buffer 0; regs hold tile 1 (if any).
  v16h ra = *(const v16h*)gA;
  v16h rb = *(const v16h*)gB;
  *(v8h*)(&sA[0][ldOff]) = __builtin_shufflevector(ra, ra, 0,1,2,3,4,5,6,7);
  *(v8h*)(&sA[0][ldOff + 8]) = __builtin_shufflevector(ra, ra, 8,9,10,11,12,13,14,15);
  *(v8h*)(&sB[0][ldOff]) = __builtin_shufflevector(rb, rb, 0,1,2,3,4,5,6,7);
  *(v8h*)(&sB[0][ldOff + 8]) = __builtin_shufflevector(rb, rb, 8,9,10,11,12,13,14,15);
  if (KT > 1) {
    gA += BK; gB += BK;
    ra = *(const v16h*)gA;
    rb = *(const v16h*)gB;
  }
  __syncthreads();

  for (int kt = 0; kt < KT; ++kt) {
    const int cur = kt & 1;
    if (kt + 1 < KT) {
      // store tile kt+1 into the other buffer (its readers finished before
      // the barrier that ended iteration kt-1)
      *(v8h*)(&sA[cur ^ 1][ldOff]) = __builtin_shufflevector(ra, ra, 0,1,2,3,4,5,6,7);
      *(v8h*)(&sA[cur ^ 1][ldOff + 8]) = __builtin_shufflevector(ra, ra, 8,9,10,11,12,13,14,15);
      *(v8h*)(&sB[cur ^ 1][ldOff]) = __builtin_shufflevector(rb, rb, 0,1,2,3,4,5,6,7);
      *(v8h*)(&sB[cur ^ 1][ldOff + 8]) = __builtin_shufflevector(rb, rb, 8,9,10,11,12,13,14,15);
      if (kt + 2 < KT) {
        gA += BK; gB += BK;
        __builtin_prefetch((const void*)(gA + BK), 0, 3);
        __builtin_prefetch((const void*)(gB + BK), 0, 3);
        ra = *(const v16h*)gA;
        rb = *(const v16h*)gB;
      }
    }

    // A fragment (16x32 f16): lanes 0-15 hold K 0..7 & 16..23, lanes 16-31 hold 8..15 & 24..31
    v16h af[2];
#pragma unroll
    for (int i = 0; i < 2; ++i) {
      const int row = wm * 32 + i * 16 + l16;
      v8h lo = *(const v8h*)(&sA[cur][row * BKP + half * 8]);
      v8h hi = *(const v8h*)(&sA[cur][row * BKP + 16 + half * 8]);
      af[i] = __builtin_shufflevector(lo, hi, 0,1,2,3,4,5,6,7,8,9,10,11,12,13,14,15);
    }
    // B fragment (32x16 f16, loaded from [N,K] layout): lanes 0-15 K 0..7 & 16..23, lanes 16-31 K 8..15 & 24..31
    v16h bf[4];
#pragma unroll
    for (int j = 0; j < 4; ++j) {
      const int row = wn * 64 + j * 16 + l16;
      v8h lo = *(const v8h*)(&sB[cur][row * BKP + half * 8]);
      v8h hi = *(const v8h*)(&sB[cur][row * BKP + 16 + half * 8]);
      bf[j] = __builtin_shufflevector(lo, hi, 0,1,2,3,4,5,6,7,8,9,10,11,12,13,14,15);
    }
#pragma unroll
    for (int i = 0; i < 2; ++i)
#pragma unroll
      for (int j = 0; j < 4; ++j)
        acc[i][j] = __builtin_amdgcn_wmma_f32_16x16x32_f16(
            false, af[i], false, bf[j], (short)0, acc[i][j], false, false);

    if (kt + 1 < KT) __syncthreads();
  }

  // Epilogue. C/D layout: VGPR r, lane l -> m = r + 8*(l>=16), n = l&15.
#pragma unroll
  for (int i = 0; i < 2; ++i) {
#pragma unroll
    for (int j = 0; j < 4; ++j) {
      const int gn = bNr + wn * 64 + j * 16 + l16;
      float bv = 0.0f;
      if (EPI == EPI_BIAS_QACT) bv = bias[gn];
#pragma unroll
      for (int r = 0; r < 8; ++r) {
        const int gm = bMr + wm * 32 + i * 16 + half * 8 + r;
        float v = acc[i][j][r];
        if (EPI == EPI_BIAS_QACT) v = qact(v + bv);
        if (EPI == EPI_F32) C32[(size_t)gm * N + gn] = v;
        else                C16[(size_t)gm * N + gn] = (_Float16)v;
      }
    }
  }
}

// ---------------------------------------------------------------------------
// Prep / epilogue helper kernels
// ---------------------------------------------------------------------------
__global__ void k_mean_phases(const float* __restrict__ qp, float* __restrict__ p,
                              int I, int Q) {
  int j = blockIdx.x * blockDim.x + threadIdx.x;
  if (j < I) {
    float s = 0.0f;
    for (int q = 0; q < Q; ++q) s += qp[j * Q + q];
    p[j] = s / (float)Q;
  }
}

__global__ void k_cos_embed(const float* __restrict__ x, const float* __restrict__ p,
                            _Float16* __restrict__ xc, int Imask) {
  size_t idx = (size_t)blockIdx.x * blockDim.x + threadIdx.x;
  int j = (int)(idx & (size_t)Imask);
  xc[idx] = (_Float16)__cosf(x[idx] * p[j]);
}

__global__ void k_cvt(const float* __restrict__ in, _Float16* __restrict__ out) {
  size_t i = (size_t)blockIdx.x * blockDim.x + threadIdx.x;
  out[i] = (_Float16)in[i];
}

// f32 [R,C] -> f16 [C,R] tiled transpose
__global__ void k_cvt_t(const float* __restrict__ in, _Float16* __restrict__ out,
                        int R, int C) {
  __shared__ float tile[32][33];
  const int bx = blockIdx.x * 32;  // C
  const int by = blockIdx.y * 32;  // R
  const int tx = threadIdx.x;      // 0..31
  const int ty = threadIdx.y;      // 0..7
  for (int i = ty; i < 32; i += 8)
    tile[i][tx] = in[(size_t)(by + i) * C + bx + tx];
  __syncthreads();
  for (int i = ty; i < 32; i += 8)
    out[(size_t)(bx + i) * R + by + tx] = (_Float16)tile[tx][i];
}

// in-place row L2-normalize (H == 1024, 256 threads, f32 accumulation)
__global__ void k_rownorm(_Float16* __restrict__ h, int H) {
  const int row = blockIdx.x;
  const int tid = threadIdx.x;
  _Float16* p = h + (size_t)row * H;
  float v[4];
  float ss = 0.0f;
#pragma unroll
  for (int i = 0; i < 4; ++i) {
    v[i] = (float)p[tid + i * 256];
    ss += v[i] * v[i];
  }
  __shared__ float red[256];
  red[tid] = ss;
  __syncthreads();
  for (int s = 128; s > 0; s >>= 1) {
    if (tid < s) red[tid] += red[tid + s];
    __syncthreads();
  }
  const float inv = 1.0f / (sqrtf(red[0]) + 1e-8f);
#pragma unroll
  for (int i = 0; i < 4; ++i) p[tid + i * 256] = (_Float16)(v[i] * inv);
}

// in-place row softmax over Ocols f32
__global__ void k_softmax(float* __restrict__ out, int Ocols) {
  const int row = blockIdx.x;
  const int tid = threadIdx.x;
  float* p = out + (size_t)row * Ocols;
  __shared__ float red[256];

  float mx = -1e30f;
  for (int c = tid; c < Ocols; c += 256) mx = fmaxf(mx, p[c]);
  red[tid] = mx;
  __syncthreads();
  for (int s = 128; s > 0; s >>= 1) {
    if (tid < s) red[tid] = fmaxf(red[tid], red[tid + s]);
    __syncthreads();
  }
  mx = red[0];
  __syncthreads();

  float sum = 0.0f;
  for (int c = tid; c < Ocols; c += 256) {
    float e = __expf(p[c] - mx);
    p[c] = e;
    sum += e;
  }
  red[tid] = sum;
  __syncthreads();
  for (int s = 128; s > 0; s >>= 1) {
    if (tid < s) red[tid] += red[tid + s];
    __syncthreads();
  }
  const float inv = 1.0f / red[0];
  for (int c = tid; c < Ocols; c += 256) p[c] *= inv;
}

// ---------------------------------------------------------------------------
extern "C" void kernel_launch(void* const* d_in, const int* in_sizes, int n_in,
                              void* d_out, int out_size, void* d_ws, size_t ws_size,
                              hipStream_t stream) {
  (void)in_sizes; (void)n_in; (void)out_size; (void)ws_size;

  const float* x  = (const float*)d_in[0];
  const float* qw = (const float*)d_in[1];
  const float* qp = (const float*)d_in[2];
  const float* w0 = (const float*)d_in[3];
  const float* b0 = (const float*)d_in[4];
  const float* e0 = (const float*)d_in[5];
  const float* w1 = (const float*)d_in[6];
  const float* b1 = (const float*)d_in[7];
  const float* e1 = (const float*)d_in[8];
  const float* mo = (const float*)d_in[9];
  float* out = (float*)d_out;

  const int B = 8192, I = 512, H = 1024, O = 4096, Q = 10;

  char* ws = (char*)d_ws;
  size_t off = 0;
  auto alloc = [&](size_t bytes) -> void* {
    void* p = ws + off;
    off = (off + bytes + 255) & ~(size_t)255;
    return p;
  };
  _Float16* XC  = (_Float16*)alloc((size_t)B * I * 2);  // cos(x*p)           [B,I]
  _Float16* WQT = (_Float16*)alloc((size_t)H * I * 2);  // q_weights^T        [H,I]
  _Float16* W0F = (_Float16*)alloc((size_t)H * H * 2);  // w0 (already [N,K])
  _Float16* E0T = (_Float16*)alloc((size_t)H * H * 2);  // e0^T               [N,K]
  _Float16* W1F = (_Float16*)alloc((size_t)H * H * 2);
  _Float16* E1T = (_Float16*)alloc((size_t)H * H * 2);
  _Float16* MT  = (_Float16*)alloc((size_t)O * H * 2);  // m_out^T            [O,H]
  _Float16* H0  = (_Float16*)alloc((size_t)B * H * 2);  // activation ping
  _Float16* H1  = (_Float16*)alloc((size_t)B * H * 2);  // activation pong
  float*    P   = (float*)alloc((size_t)I * 4);         // mean phases

  // --- operand preparation (f32 -> f16, transposes into [N,K] layout) ---
  k_mean_phases<<<(I + 255) / 256, 256, 0, stream>>>(qp, P, I, Q);
  k_cos_embed<<<(B * I) / 256, 256, 0, stream>>>(x, P, XC, I - 1);
  dim3 tb(32, 8);
  k_cvt_t<<<dim3(H / 32, I / 32), tb, 0, stream>>>(qw, WQT, I, H);
  k_cvt<<<(H * H) / 256, 256, 0, stream>>>(w0, W0F);
  k_cvt_t<<<dim3(H / 32, H / 32), tb, 0, stream>>>(e0, E0T, H, H);
  k_cvt<<<(H * H) / 256, 256, 0, stream>>>(w1, W1F);
  k_cvt_t<<<dim3(H / 32, H / 32), tb, 0, stream>>>(e1, E1T, H, H);
  k_cvt_t<<<dim3(O / 32, H / 32), tb, 0, stream>>>(mo, MT, H, O);

  // --- WMMA GEMM pipeline ---
  dim3 gH(H / BN, B / BM);  // 8 x 64 blocks
  // h = cos(x*p) @ q_weights
  k_gemm<EPI_F16><<<gH, 256, 0, stream>>>(XC, WQT, nullptr, H0, nullptr, B, H, I);
  // layer 0: qact(h @ w0.T + b0)
  k_gemm<EPI_BIAS_QACT><<<gH, 256, 0, stream>>>(H0, W0F, b0, H1, nullptr, B, H, H);
  // @ e0, then row-normalize
  k_gemm<EPI_F16><<<gH, 256, 0, stream>>>(H1, E0T, nullptr, H0, nullptr, B, H, H);
  k_rownorm<<<B, 256, 0, stream>>>(H0, H);
  // layer 1
  k_gemm<EPI_BIAS_QACT><<<gH, 256, 0, stream>>>(H0, W1F, b1, H1, nullptr, B, H, H);
  k_gemm<EPI_F16><<<gH, 256, 0, stream>>>(H1, E1T, nullptr, H0, nullptr, B, H, H);
  k_rownorm<<<B, 256, 0, stream>>>(H0, H);
  // logits = h @ m_out -> d_out (f32), softmax in-place
  dim3 gO(O / BN, B / BM);  // 32 x 64 blocks
  k_gemm<EPI_F32><<<gO, 256, 0, stream>>>(H0, MT, nullptr, nullptr, out, B, O, H);
  k_softmax<<<B, 256, 0, stream>>>(out, O);
}
